// KANTransformerBlock_12506944766033
// MI455X (gfx1250) — matrix operand order repaired and checked by reading, not compiled, not run on hardware
//
#include <hip/hip_runtime.h>
#include <math.h>

typedef __bf16 bf16;
typedef __attribute__((ext_vector_type(8)))  bf16  v8bf;
typedef __attribute__((ext_vector_type(16))) bf16  v16bf;
typedef __attribute__((ext_vector_type(8)))  float v8f;

union V16BF { v16bf v; v8bf h[2]; };

// ---------------------------------------------------------------------------
// Cox-de Boor cubic B-spline basis on uniform knots over [-1,1], G intervals.
// Produces G+3 basis values (k = 3), matching the reference exactly.
// ---------------------------------------------------------------------------
template <int G>
__device__ inline void bspline_eval(float x, float* out) {
    constexpr int K = 3;
    const float h = 2.0f / (float)G;
    float b[G + 2 * K];
#pragma unroll
    for (int j = 0; j < G + 2 * K; ++j) {
        float p0 = ((float)(j - K)) * h - 1.0f;
        float p1 = ((float)(j + 1 - K)) * h - 1.0f;
        b[j] = (x >= p0 && x < p1) ? 1.0f : 0.0f;
    }
#pragma unroll
    for (int p = 1; p <= K; ++p) {
        const float invph = 1.0f / ((float)p * h);
#pragma unroll
        for (int j = 0; j < G + 2 * K - p; ++j) {
            float t0  = ((float)(j - K)) * h - 1.0f;
            float t1p = ((float)(j + 1 + p - K)) * h - 1.0f;
            b[j] = (x - t0) * invph * b[j] + (t1p - x) * invph * b[j + 1];
        }
    }
#pragma unroll
    for (int j = 0; j < G + K; ++j) out[j] = b[j];
}

// ---------------------------------------------------------------------------
// Interleave base + spline weights into bf16 W[o][i*(SPL+1)+g], slot 0 = base.
// ---------------------------------------------------------------------------
__global__ void pack_weights(const float* __restrict__ wb,
                             const float* __restrict__ wsp,
                             bf16* __restrict__ W, int SPL, int total) {
    int idx = blockIdx.x * blockDim.x + threadIdx.x;
    if (idx >= total) return;
    const int GP = SPL + 1;
    bf16* dst = W + (size_t)idx * GP;
    dst[0] = (bf16)wb[idx];
    const float* src = wsp + (size_t)idx * SPL;
    for (int g = 0; g < SPL; ++g) dst[1 + g] = (bf16)src[g];
}

// ---------------------------------------------------------------------------
// Expand activations once per layer: PHI[tok][i*(G+4)+g], slot 0 = silu(x),
// slots 1..G+3 = B-spline bases. Flat: idx = tok*DIN + i -> PHI + idx*(G+4).
// ---------------------------------------------------------------------------
template <int G>
__global__ __launch_bounds__(256)
void kan_expand(const float* __restrict__ X, bf16* __restrict__ PHI, int total) {
    int idx = blockIdx.x * blockDim.x + threadIdx.x;
    if (idx >= total) return;
    constexpr int GP4 = G + 4;
    const float xv = X[idx];
    bf16* dst = PHI + (size_t)idx * GP4;
    dst[0] = (bf16)(xv / (1.0f + __expf(-xv)));  // silu base path
    float bs[G + 3];
    bspline_eval<G>(xv, bs);
#pragma unroll
    for (int g = 0; g < G + 3; ++g) dst[1 + g] = (bf16)bs[g];
}

// ---------------------------------------------------------------------------
// bf16 WMMA GEMM:  OUT[t, o] = epilogue( sum_k PHI[t, k] * W[o, k] ).
//   K     : expanded reduction length (= DIN*(G+4)), multiple of 64
//   MODE 0: OUT = RES * sigmoid(acc)   (attention gate, RES = x)
//   MODE 1: OUT = gelu_exact(acc)      (FFN mid)
//   MODE 2: OUT = RES + acc            (FFN out + residual, RES = xg)
// Workgroup: 256 threads = 8 waves; tile 128 tokens x 128 outputs.
// Wave (mi = w&3, ni = w>>2) owns a 32x64 block = 2x4 WMMA C tiles
// (8 WMMAs per 32-k substep against 6 fragment loads).
// Direct global->LDS staging (transient registers only -> no scratch spill);
// cross-workgroup residency (32 KB LDS/WG) hides the staging latency, and
// next-chunk prefetches keep the staging loads in cache.
// ---------------------------------------------------------------------------
template <int K, int DOUT, int MODE>
__global__ __launch_bounds__(256)
void kan_gemm(const bf16* __restrict__ PHI,   // [TC, K] chunk-local
              const bf16* __restrict__ W,     // [DOUT, K]
              float* __restrict__ OUT,        // [TC, DOUT] at chunk base
              const float* __restrict__ RES) {// [TC, DOUT] at chunk base / null
    constexpr int KC  = 64;        // LDS K chunk (bf16 elements)
    constexpr int NCH = K / KC;

    __shared__ bf16 As[128 * KC];  // [token][k]  (16 KB)
    __shared__ bf16 Bs[128 * KC];  // [out][k]    (16 KB)

    const int tid   = threadIdx.x;
    const int lane  = tid & 31;
    const int wave  = tid >> 5;
    const int mi    = wave & 3;    // 4 m-waves x 32 rows
    const int ni    = wave >> 2;   // 2 n-waves x 64 cols
    const int mBase = blockIdx.x * 128;
    const int nBase = blockIdx.y * 128;

    v8f acc[2][4] = {};

    for (int c = 0; c < NCH; ++c) {
        __syncthreads();  // previous chunk's LDS reads complete
        // stage chunk c: 128 rows x 64 halves per tile = 1024 uint4 each;
        // 4 x uint4 per thread per tile, transient registers only.
        const size_t kg = (size_t)c * KC;
#pragma unroll
        for (int q = 0; q < 4; ++q) {
            const int i   = tid + 256 * q;
            const int row = i >> 3;   // 0..127
            const int cv  = i & 7;    // 0..7 (x 8 halves)
            *reinterpret_cast<uint4*>(&As[row * KC + cv * 8]) =
                *reinterpret_cast<const uint4*>(
                    &PHI[(size_t)(mBase + row) * K + kg + cv * 8]);
            *reinterpret_cast<uint4*>(&Bs[row * KC + cv * 8]) =
                *reinterpret_cast<const uint4*>(
                    &W[(size_t)(nBase + row) * K + kg + cv * 8]);
        }
        if (c + 1 < NCH) {  // global_prefetch_b8 one chunk ahead
            const size_t kn = kg + KC;
            __builtin_prefetch(&W[(size_t)(nBase + tid / 2) * K + kn], 0, 1);
            __builtin_prefetch(&PHI[(size_t)(mBase + tid / 2) * K + kn], 0, 1);
        }
        __syncthreads();  // LDS chunk visible

        // WMMA fragment addressing per CDNA5 16-bit layouts
        const int asel = (lane >= 16) ? 8 : 0;    // A: hi half-wave holds K+8
        const int bsel = (lane >= 16) ? 8 : 0;    // B: hi half-wave holds K+8
        const int arow = mi * 32 + (lane & 15);
        const int brow = ni * 64 + (lane & 15);
#pragma unroll
        for (int kk = 0; kk < KC / 32; ++kk) {
            const int ko = kk * 32;
            V16BF a[2], b[4];
#pragma unroll
            for (int mt = 0; mt < 2; ++mt) {
                const bf16* ap = &As[(arow + mt * 16) * KC + ko + asel];
                a[mt].h[0] = *reinterpret_cast<const v8bf*>(ap);
                a[mt].h[1] = *reinterpret_cast<const v8bf*>(ap + 16);
            }
#pragma unroll
            for (int nt = 0; nt < 4; ++nt) {
                const bf16* bp = &Bs[(brow + nt * 16) * KC + ko + bsel];
                b[nt].h[0] = *reinterpret_cast<const v8bf*>(bp);
                b[nt].h[1] = *reinterpret_cast<const v8bf*>(bp + 16);
            }
#pragma unroll
            for (int mt = 0; mt < 2; ++mt) {
#pragma unroll
                for (int nt = 0; nt < 4; ++nt) {
                    acc[mt][nt] = __builtin_amdgcn_wmma_f32_16x16x32_bf16(
                        false, a[mt].v, false, b[nt].v, (short)0, acc[mt][nt],
                        false, false);
                }
            }
        }
    }

    // fused epilogue (C layout: VGPR r -> row r + 8*(lane>=16), col = lane%16)
#pragma unroll
    for (int mt = 0; mt < 2; ++mt) {
#pragma unroll
        for (int nt = 0; nt < 4; ++nt) {
            const int col  = nBase + ni * 64 + nt * 16 + (lane & 15);
            const int rowb = mBase + mi * 32 + mt * 16 + ((lane >= 16) ? 8 : 0);
#pragma unroll
            for (int r = 0; r < 8; ++r) {
                const size_t o = (size_t)(rowb + r) * DOUT + col;
                const float v = acc[mt][nt][r];
                if (MODE == 0) {
                    OUT[o] = RES[o] / (1.0f + __expf(-v));   // x * sigmoid(kan)
                } else if (MODE == 1) {
                    OUT[o] = 0.5f * v * (1.0f + erff(v * 0.70710678118654752f));
                } else {
                    OUT[o] = RES[o] + v;                     // z = xg + y
                }
            }
        }
    }
}

// ---------------------------------------------------------------------------
// Row-wise LayerNorm in place over D columns. One 256-thread block per row.
// ---------------------------------------------------------------------------
__global__ __launch_bounds__(256)
void layernorm_inplace(float* __restrict__ Z, const float* __restrict__ w,
                       const float* __restrict__ b, int D) {
    float* z = Z + (size_t)blockIdx.x * D;
    float s = 0.0f, s2 = 0.0f;
    for (int i = threadIdx.x; i < D; i += 256) {
        const float v = z[i];
        s += v;
        s2 += v * v;
    }
    __shared__ float r0[256], r1[256];
    r0[threadIdx.x] = s;
    r1[threadIdx.x] = s2;
    __syncthreads();
    for (int st = 128; st > 0; st >>= 1) {
        if (threadIdx.x < st) {
            r0[threadIdx.x] += r0[threadIdx.x + st];
            r1[threadIdx.x] += r1[threadIdx.x + st];
        }
        __syncthreads();
    }
    const float mu  = r0[0] / (float)D;
    const float var = r1[0] / (float)D - mu * mu;
    const float inv = rsqrtf(var + 1e-5f);
    for (int i = threadIdx.x; i < D; i += 256) {
        z[i] = (z[i] - mu) * inv * w[i] + b[i];
    }
}

// ---------------------------------------------------------------------------
// Launch. Inputs (setup_inputs order):
//  0 x[16,512,512] 1 w_base_attn[512,512] 2 w_spline_attn[512,512,8]
//  3 w_base_f1[1024,512] 4 w_spline_f1[1024,512,6]
//  5 w_base_f2[512,1024] 6 w_spline_f2[512,1024,6] 7 ln_w[512] 8 ln_b[512]
// Workspace (~128 MB): bf16 packed weights (19.4 MB) + PHI chunk buffer
// (4096 x 7168 bf16 = 58.7 MB) + xg (16.8 MB) + h (33.6 MB).
// ---------------------------------------------------------------------------
extern "C" void kernel_launch(void* const* d_in, const int* in_sizes, int n_in,
                              void* d_out, int out_size, void* d_ws, size_t ws_size,
                              hipStream_t stream) {
    const float* x   = (const float*)d_in[0];
    const float* wba = (const float*)d_in[1];
    const float* wsa = (const float*)d_in[2];
    const float* wb1 = (const float*)d_in[3];
    const float* ws1 = (const float*)d_in[4];
    const float* wb2 = (const float*)d_in[5];
    const float* ws2 = (const float*)d_in[6];
    const float* lnw = (const float*)d_in[7];
    const float* lnb = (const float*)d_in[8];
    float* out = (float*)d_out;

    const int D = 512, H = 1024;
    const int NT = in_sizes[0] / D;  // 8192 tokens
    const int TC = 4096;             // token chunk (bounds PHI workspace)

    // workspace layout
    bf16* Wattn = (bf16*)d_ws;                          // 512  * 4608 bf16
    bf16* Wf1   = Wattn + (size_t)D * (D * 9);          // 1024 * 3584 bf16
    bf16* Wf2   = Wf1 + (size_t)H * (D * 7);            // 512  * 7168 bf16
    bf16* PHI   = Wf2 + (size_t)D * (H * 7);            // TC * 7168 bf16 (max)
    float* xg   = (float*)(PHI + (size_t)TC * (H * 7)); // [NT, 512]
    float* hbuf = xg + (size_t)NT * D;                  // [NT, 1024]

    // 1) pack weights to bf16 interleaved [o][i][slot]
    pack_weights<<<(D * D + 255) / 256, 256, 0, stream>>>(wba, wsa, Wattn, 8, D * D);
    pack_weights<<<(H * D + 255) / 256, 256, 0, stream>>>(wb1, ws1, Wf1, 6, H * D);
    pack_weights<<<(D * H + 255) / 256, 256, 0, stream>>>(wb2, ws2, Wf2, 6, D * H);

    // 2) attention gate: xg = x * sigmoid(KAN_attn(x))   (G=5 -> K=4608)
    for (int t0 = 0; t0 < NT; t0 += TC) {
        kan_expand<5><<<(TC * D + 255) / 256, 256, 0, stream>>>(
            x + (size_t)t0 * D, PHI, TC * D);
        kan_gemm<4608, 512, 0><<<dim3(TC / 128, D / 128), 256, 0, stream>>>(
            PHI, Wattn, xg + (size_t)t0 * D, x + (size_t)t0 * D);
    }

    // 3) h = gelu(KAN_f1(xg))                            (G=3 -> K=3584)
    for (int t0 = 0; t0 < NT; t0 += TC) {
        kan_expand<3><<<(TC * D + 255) / 256, 256, 0, stream>>>(
            xg + (size_t)t0 * D, PHI, TC * D);
        kan_gemm<3584, 1024, 1><<<dim3(TC / 128, H / 128), 256, 0, stream>>>(
            PHI, Wf1, hbuf + (size_t)t0 * H, nullptr);
    }

    // 4) z = xg + KAN_f2(h) -> d_out                     (G=3 -> K=7168)
    for (int t0 = 0; t0 < NT; t0 += TC) {
        kan_expand<3><<<(TC * H + 255) / 256, 256, 0, stream>>>(
            hbuf + (size_t)t0 * H, PHI, TC * H);
        kan_gemm<7168, 512, 2><<<dim3(TC / 128, D / 128), 256, 0, stream>>>(
            PHI, Wf2, out + (size_t)t0 * D, xg + (size_t)t0 * D);
    }

    // 5) LayerNorm in place
    layernorm_inplace<<<NT, 256, 0, stream>>>(out, lnw, lnb, D);
}
